// Decoder_62680752717853
// MI455X (gfx1250) — matrix operand order repaired and checked by reading, not compiled, not run on hardware
//
#include <hip/hip_runtime.h>
#include <hip/hip_bf16.h>
#include <math.h>

typedef __attribute__((ext_vector_type(16))) __bf16 v16bf;
typedef __attribute__((ext_vector_type(8)))  float  v8f;

#define S_LEN 1024
#define BATCH 64
#define HDIM  512
#define ADIM  512
#define EDIM  256
#define FDIM  32
#define VDIM  5000
#define G4H   2048
#define NPAD  5120

__device__ __forceinline__ unsigned short f2bf(float f) {
  unsigned int u = __float_as_uint(f);
  unsigned int r = (u + 0x7FFFu + ((u >> 16) & 1u)) >> 16;   // round-to-nearest-even
  return (unsigned short)r;
}
__device__ __forceinline__ __bf16 u2bf(unsigned short s) {
  union { unsigned short u; __bf16 b; } v; v.u = s; return v.b;
}
__device__ __forceinline__ float sigf(float x) { return 1.0f / (1.0f + expf(-x)); }

// ---------------- small scalar kernels ----------------

__global__ void k_embed(const int* __restrict__ idx, const float* __restrict__ emb,
                        float* __restrict__ x) {
  int b = blockIdx.x, e = threadIdx.x;
  x[b * EDIM + e] = emb[(size_t)idx[b] * EDIM + e];
}

__global__ void k_gates(const float* __restrict__ X, int K1, const float* __restrict__ Wih,
                        const float* __restrict__ Hst, int K2, const float* __restrict__ Whh,
                        const float* __restrict__ bih, const float* __restrict__ bhh,
                        float* __restrict__ g) {
  int u = blockIdx.x, b = threadIdx.x;          // grid 2048, block 64
  float acc = bih[u] + bhh[u];
  const float* w  = Wih + (size_t)u * K1;
  const float* xr = X   + (size_t)b * K1;
  for (int k = 0; k < K1; ++k) acc += xr[k] * w[k];
  w  = Whh + (size_t)u * K2;
  xr = Hst + (size_t)b * K2;
  for (int k = 0; k < K2; ++k) acc += xr[k] * w[k];
  g[(size_t)b * G4H + u] = acc;
}

__global__ void k_cell(const float* __restrict__ g, const float* __restrict__ c0,
                       float* __restrict__ hout, float* __restrict__ cout,
                       float* __restrict__ hws) {
  int t = blockIdx.x * blockDim.x + threadIdx.x;   // 0..32767
  int b = t >> 9, h = t & 511;
  const float* gr = g + (size_t)b * G4H;
  float i = gr[h], f = gr[512 + h], gg = gr[1024 + h], o = gr[1536 + h];
  float c  = sigf(f) * c0[t] + sigf(i) * tanhf(gg);
  float hh = sigf(o) * tanhf(c);
  cout[t] = c; hout[t] = hh; hws[t] = hh;
}

__global__ void k_decp(const float* __restrict__ dec, const float* __restrict__ W1w,
                       const float* __restrict__ W1b, float* __restrict__ decp) {
  int a = blockIdx.x, b = threadIdx.x;             // grid 512, block 64
  float acc = W1b[a];
  const float* w = W1w + (size_t)a * HDIM;
  const float* x = dec + (size_t)b * HDIM;
  for (int k = 0; k < HDIM; ++k) acc += x[k] * w[k];
  decp[(size_t)b * ADIM + a] = acc;
}

__global__ void k_conv(const float* __restrict__ aw, const float* __restrict__ cw,
                       const float* __restrict__ cb, float* __restrict__ convsbf) {
  int s = blockIdx.x;                              // grid 1024, block 256
  for (int p = threadIdx.x; p < BATCH * FDIM; p += blockDim.x) {
    int b = p >> 5, f = p & 31;
    float acc = cb[f];
#pragma unroll
    for (int k = 0; k < 3; ++k) {
      int sp = s + k - 1;
      if (sp >= 0 && sp < S_LEN) acc += aw[b * S_LEN + sp] * cw[f * 3 + k];
    }
    convsbf[((size_t)s * BATCH + b) * FDIM + f] = acc;
  }
}

// ---------------- fragment packing (B-matrix layout, contiguous k = 16*half + j) ----------------
// B 32x16 bf16: lanes 0-15 hold K=0..15, lanes 16-31 hold K=16..31; lane carries col n.
// frag[((ktile*2 + half)*N + n)*16 + j] = W[n][ktile*32 + 16*half + j]

__global__ void k_prep_w2frag(const float* __restrict__ W2w, unsigned short* __restrict__ frag) {
  int k = blockIdx.x, n = threadIdx.x;             // 512 x 512
  int ktile = k >> 5, w = k & 31;
  int half = w >> 4, j = w & 15;
  frag[(((size_t)(ktile * 2 + half)) * ADIM + n) * 16 + j] = f2bf(W2w[(size_t)n * HDIM + k]);
}

__global__ void k_prep_outwfrag(const float* __restrict__ OW, unsigned short* __restrict__ frag) {
  int k = blockIdx.x;                                  // 0..1023
  int n = blockIdx.y * blockDim.x + threadIdx.x;       // 0..5119
  int ktile = k >> 5, w = k & 31;
  int half = w >> 4, j = w & 15;
  unsigned short v = (n < VDIM) ? f2bf(OW[(size_t)n * 1024 + k]) : (unsigned short)0;
  frag[(((size_t)(ktile * 2 + half)) * NPAD + n) * 16 + j] = v;
}

__global__ void k_prep_bias(const float* w2b, const float* w3b, const float* attb,
                            float* biasA) {
  int n = threadIdx.x;
  biasA[n] = w2b[n] + w3b[n] + attb[n];
}

// ---------------- fused attention-energy GEMM (the 34 GFLOP op) ----------------
// Block: 16 (s,b) rows x 512 A-cols, 8 waves, each wave 4 16x16 tiles.
// A panel (16x512 bf16, 16KB) staged in LDS ONCE; B fragments double-buffered in regs
// with compile-time indices (no pointer swap -> no accumulator moves).

__global__ __launch_bounds__(256) void k_attn_score(
    const float* __restrict__ enc, const unsigned short* __restrict__ w2frag,
    const float* __restrict__ decp, const float* __restrict__ convsbf,
    const float* __restrict__ W3w, const float* __restrict__ biasA,
    const float* __restrict__ vvec, float* __restrict__ scores) {
  __shared__ unsigned int ApanU[16 * 256];   // 16 rows x 512 bf16 (packed pairs)
  __shared__ float convS[16 * 32];
  __shared__ float sred[16];
  const int tid  = threadIdx.x;
  const int wave = tid >> 5, lane = tid & 31;
  const int half = lane >> 4, lm = lane & 15;
  const size_t rowbase = (size_t)blockIdx.x * 16;

  if (tid < 16) sred[tid] = 0.0f;
  for (int p = tid; p < 16 * 32; p += 256)
    convS[p] = convsbf[(rowbase + (p >> 5)) * FDIM + (p & 31)];

  // stage full A panel once: 2048 float4 loads, f32 -> packed bf16 pairs
  const float4* encv = reinterpret_cast<const float4*>(enc + rowbase * HDIM);
  for (int p = tid; p < 2048; p += 256) {
    float4 x = encv[p];
    ApanU[p * 2]     = (unsigned)f2bf(x.x) | ((unsigned)f2bf(x.y) << 16);
    ApanU[p * 2 + 1] = (unsigned)f2bf(x.z) | ((unsigned)f2bf(x.w) << 16);
  }
  __syncthreads();
  const unsigned short* Apan = reinterpret_cast<const unsigned short*>(ApanU);

  auto loadB = [&](int ktile, v16bf* dst) {
#pragma unroll
    for (int t = 0; t < 4; ++t) {
      const int n = wave * 64 + t * 16 + lm;
      dst[t] = *reinterpret_cast<const v16bf*>(
          &w2frag[(((size_t)(ktile * 2 + half)) * ADIM + n) * 16]);
    }
  };
  auto loadA = [&](int ktile) -> v16bf {
    v16bf a;
    const unsigned short* ap = &Apan[lm * 512 + ktile * 32 + 16 * half];
#pragma unroll
    for (int j = 0; j < 8; ++j) { a[j] = u2bf(ap[j]); a[j + 8] = u2bf(ap[8 + j]); }
    return a;
  };

  v8f acc[4] = {};
  v16bf Ba[4], Bb[4];
  loadB(0, Ba);
  for (int kt = 0; kt < 16; kt += 2) {
    loadB(kt + 1, Bb);
    {
      const v16bf a = loadA(kt);
#pragma unroll
      for (int t = 0; t < 4; ++t)
        acc[t] = __builtin_amdgcn_wmma_f32_16x16x32_bf16(
            false, a, false, Ba[t], (short)0, acc[t], false, false);
    }
    if (kt + 2 < 16) loadB(kt + 2, Ba);
    {
      const v16bf a = loadA(kt + 1);
#pragma unroll
      for (int t = 0; t < 4; ++t)
        acc[t] = __builtin_amdgcn_wmma_f32_16x16x32_bf16(
            false, a, false, Bb[t], (short)0, acc[t], false, false);
    }
  }

  // epilogue: e = tanh(enc_p + dec_p + conv_p + bias); partial score = sum_n v[n]*e
  float part[8] = {0, 0, 0, 0, 0, 0, 0, 0};
#pragma unroll
  for (int t = 0; t < 4; ++t) {
    const int n = wave * 64 + t * 16 + lm;
    const float bn = biasA[n];
    const float vn = vvec[n];
    const float* w3n = &W3w[n * FDIM];
    float cv[8] = {0, 0, 0, 0, 0, 0, 0, 0};
    for (int f = 0; f < FDIM; ++f) {
      const float wv = w3n[f];
#pragma unroll
      for (int j = 0; j < 8; ++j) cv[j] += convS[(8 * half + j) * 32 + f] * wv;
    }
#pragma unroll
    for (int j = 0; j < 8; ++j) {
      const int m = 8 * half + j;
      const int b = (int)((rowbase + m) & 63);
      const float e = tanhf(acc[t][j] + decp[(size_t)b * ADIM + n] + bn + cv[j]);
      part[j] += vn * e;
    }
  }
#pragma unroll
  for (int off = 8; off >= 1; off >>= 1) {
#pragma unroll
    for (int j = 0; j < 8; ++j) part[j] += __shfl_xor(part[j], off, 32);
  }
  if (lm == 0) {
#pragma unroll
    for (int j = 0; j < 8; ++j) atomicAdd(&sred[8 * half + j], part[j]);
  }
  __syncthreads();
  if (tid < 16) {
    const size_t row = rowbase + tid;
    const int s = (int)(row >> 6), b = (int)(row & 63);
    scores[(size_t)b * S_LEN + s] = sred[tid];
  }
}

// ---------------- softmax / context / output GEMM / log-softmax ----------------

__global__ void k_softmax(const float* __restrict__ scores, float* __restrict__ attnws,
                          float* __restrict__ attnout) {
  const int b = blockIdx.x, tid = threadIdx.x;
  __shared__ float red[256];
  float mx = -3.4e38f;
  for (int s = tid; s < S_LEN; s += 256) mx = fmaxf(mx, scores[b * S_LEN + s]);
  red[tid] = mx; __syncthreads();
  for (int o = 128; o > 0; o >>= 1) { if (tid < o) red[tid] = fmaxf(red[tid], red[tid + o]); __syncthreads(); }
  const float M = red[0]; __syncthreads();
  float sm = 0.0f;
  for (int s = tid; s < S_LEN; s += 256) sm += expf(scores[b * S_LEN + s] - M);
  red[tid] = sm; __syncthreads();
  for (int o = 128; o > 0; o >>= 1) { if (tid < o) red[tid] += red[tid + o]; __syncthreads(); }
  const float inv = 1.0f / red[0];
  for (int s = tid; s < S_LEN; s += 256) {
    const float a = expf(scores[b * S_LEN + s] - M) * inv;
    attnws[b * S_LEN + s] = a;
    attnout[b * S_LEN + s] = a;
  }
}

__global__ void k_context(const float* __restrict__ attn, const float* __restrict__ enc,
                          const float* __restrict__ dec, unsigned short* __restrict__ concatbf) {
  const int b = blockIdx.x, h = threadIdx.x;       // grid 64, block 512
  float acc = 0.0f;
  for (int s = 0; s < S_LEN; ++s)
    acc += attn[b * S_LEN + s] * enc[((size_t)s * BATCH + b) * HDIM + h];
  concatbf[(size_t)b * 1024 + h]       = f2bf(acc);
  concatbf[(size_t)b * 1024 + 512 + h] = f2bf(dec[(size_t)b * HDIM + h]);
}

// Output GEMM: A panel (16x1024 bf16, 32KB) staged via ASYNC global->LDS copy,
// B fragments double-buffered in regs with compile-time indices.
__global__ __launch_bounds__(256) void k_out_gemm(
    const unsigned short* __restrict__ inbf, const unsigned short* __restrict__ wfrag,
    const float* __restrict__ outb, float* __restrict__ logits) {
  __shared__ unsigned short Apan[16 * 1024];       // 32KB
  const int tid  = threadIdx.x;
  const int wave = tid >> 5, lane = tid & 31;
  const int half = lane >> 4, lm = lane & 15;
  const int rowbase = blockIdx.y * 16;
  const int nbase   = blockIdx.x * 512;

  // async stage: 2048 x b128 lane-transfers (8 per thread), tracked by ASYNCcnt
  {
    const unsigned short* src = inbf + (size_t)rowbase * 1024;
    for (int p = tid; p < 2048; p += 256) {
      unsigned       ldsoff = (unsigned)(size_t)(&Apan[p * 8]);   // flat LDS addr[31:0]
      unsigned long long ga = (unsigned long long)(src + p * 8);
      asm volatile("global_load_async_to_lds_b128 %0, %1, off"
                   :: "v"(ldsoff), "v"(ga) : "memory");
    }
    asm volatile("s_wait_asynccnt 0x0" ::: "memory");
  }
  __syncthreads();

  auto loadB = [&](int ktile, v16bf* dst) {
#pragma unroll
    for (int t = 0; t < 4; ++t) {
      const int n = nbase + wave * 64 + t * 16 + lm;
      dst[t] = *reinterpret_cast<const v16bf*>(
          &wfrag[(((size_t)(ktile * 2 + half)) * NPAD + n) * 16]);
    }
  };
  auto loadA = [&](int ktile) -> v16bf {
    v16bf a;
    const unsigned short* ap = &Apan[lm * 1024 + ktile * 32 + 16 * half];
#pragma unroll
    for (int j = 0; j < 8; ++j) { a[j] = u2bf(ap[j]); a[j + 8] = u2bf(ap[8 + j]); }
    return a;
  };

  v8f acc[4] = {};
  v16bf Ba[4], Bb[4];
  loadB(0, Ba);
  for (int kt = 0; kt < 32; kt += 2) {
    loadB(kt + 1, Bb);
    {
      const v16bf a = loadA(kt);
#pragma unroll
      for (int t = 0; t < 4; ++t)
        acc[t] = __builtin_amdgcn_wmma_f32_16x16x32_bf16(
            false, a, false, Ba[t], (short)0, acc[t], false, false);
    }
    if (kt + 2 < 32) loadB(kt + 2, Ba);
    {
      const v16bf a = loadA(kt + 1);
#pragma unroll
      for (int t = 0; t < 4; ++t)
        acc[t] = __builtin_amdgcn_wmma_f32_16x16x32_bf16(
            false, a, false, Bb[t], (short)0, acc[t], false, false);
    }
  }
#pragma unroll
  for (int t = 0; t < 4; ++t) {
    const int n = nbase + wave * 64 + t * 16 + lm;
    if (n < VDIM) {
#pragma unroll
      for (int j = 0; j < 8; ++j) {
        const int m = 8 * half + j;
        logits[(size_t)(rowbase + m) * VDIM + n] = acc[t][j] + outb[n];
      }
    }
  }
}

__global__ void k_logsoftmax(const float* __restrict__ logits, float* __restrict__ logp) {
  const int b = blockIdx.x, tid = threadIdx.x;
  __shared__ float red[256];
  float mx = -3.4e38f;
  for (int v = tid; v < VDIM; v += 256) mx = fmaxf(mx, logits[(size_t)b * VDIM + v]);
  red[tid] = mx; __syncthreads();
  for (int o = 128; o > 0; o >>= 1) { if (tid < o) red[tid] = fmaxf(red[tid], red[tid + o]); __syncthreads(); }
  const float M = red[0]; __syncthreads();
  float sm = 0.0f;
  for (int v = tid; v < VDIM; v += 256) sm += expf(logits[(size_t)b * VDIM + v] - M);
  red[tid] = sm; __syncthreads();
  for (int o = 128; o > 0; o >>= 1) { if (tid < o) red[tid] += red[tid + o]; __syncthreads(); }
  const float lse = M + logf(red[0]);
  for (int v = tid; v < VDIM; v += 256)
    logp[(size_t)b * VDIM + v] = logits[(size_t)b * VDIM + v] - lse;
}

// ---------------- launch ----------------

extern "C" void kernel_launch(void* const* d_in, const int* in_sizes, int n_in,
                              void* d_out, int out_size, void* d_ws, size_t ws_size,
                              hipStream_t stream) {
  const int*   idx   = (const int*)  d_in[0];
  const float* h0    = (const float*)d_in[1];
  const float* c0    = (const float*)d_in[2];
  const float* enc   = (const float*)d_in[3];
  const float* aw    = (const float*)d_in[4];
  const float* emb   = (const float*)d_in[5];
  const float* wih0  = (const float*)d_in[6];
  const float* whh0  = (const float*)d_in[7];
  const float* bih0  = (const float*)d_in[8];
  const float* bhh0  = (const float*)d_in[9];
  const float* wih1  = (const float*)d_in[10];
  const float* whh1  = (const float*)d_in[11];
  const float* bih1  = (const float*)d_in[12];
  const float* bhh1  = (const float*)d_in[13];
  const float* W1w   = (const float*)d_in[14];
  const float* W1b   = (const float*)d_in[15];
  const float* W2w   = (const float*)d_in[16];
  const float* W2b   = (const float*)d_in[17];
  const float* W3w   = (const float*)d_in[18];
  const float* W3b   = (const float*)d_in[19];
  const float* convw = (const float*)d_in[20];
  const float* convb = (const float*)d_in[21];
  const float* vvec  = (const float*)d_in[22];
  const float* attb  = (const float*)d_in[23];
  const float* outw  = (const float*)d_in[24];
  const float* outb  = (const float*)d_in[25];

  float* out    = (float*)d_out;
  float* o_logp = out;               // 64*5000
  float* o_hid  = out + 320000;      // 2*64*512
  float* o_cell = out + 385536;      // 2*64*512
  float* o_attn = out + 451072;      // 64*1024

  float* ws       = (float*)d_ws;
  float* w_x      = ws + 0;          // 64*256
  float* w_g      = ws + 16384;      // 64*2048 (reused for both LSTM layers)
  float* w_h1     = ws + 147456;     // 64*512
  float* w_dec    = ws + 180224;     // 64*512
  float* w_decp   = ws + 212992;     // 64*512
  float* w_conv   = ws + 245760;     // 1024*64*32
  float* w_scores = ws + 2342912;    // 64*1024
  float* w_attn   = ws + 2408448;    // 64*1024
  float* w_logits = ws + 2473984;    // 64*5000
  float* w_bias   = ws + 2793984;    // 512
  unsigned short* u16base   = (unsigned short*)((char*)d_ws + 11177984);
  unsigned short* w_w2frag  = u16base;                       // 512*512 bf16
  unsigned short* w_outfrag = u16base + 262144;              // 1024*5120 bf16
  unsigned short* w_concat  = u16base + 262144 + 5242880;    // 64*1024 bf16

  // independent prep
  k_prep_w2frag<<<512, 512, 0, stream>>>(W2w, w_w2frag);
  k_prep_outwfrag<<<dim3(1024, 5), 1024, 0, stream>>>(outw, w_outfrag);
  k_prep_bias<<<1, 512, 0, stream>>>(W2b, W3b, attb, w_bias);
  k_conv<<<S_LEN, 256, 0, stream>>>(aw, convw, convb, w_conv);

  // LSTM chain
  k_embed<<<BATCH, EDIM, 0, stream>>>(idx, emb, w_x);
  k_gates<<<G4H, BATCH, 0, stream>>>(w_x, EDIM, wih0, h0, HDIM, whh0, bih0, bhh0, w_g);
  k_cell<<<128, 256, 0, stream>>>(w_g, c0, o_hid, o_cell, w_h1);
  k_gates<<<G4H, BATCH, 0, stream>>>(w_h1, HDIM, wih1, h0 + 32768, HDIM, whh1, bih1, bhh1, w_g);
  k_cell<<<128, 256, 0, stream>>>(w_g, c0 + 32768, o_hid + 32768, o_cell + 32768, w_dec);
  k_decp<<<ADIM, BATCH, 0, stream>>>(w_dec, W1w, W1b, w_decp);

  // fused attention energy GEMM + scoring
  k_attn_score<<<(S_LEN * BATCH) / 16, 256, 0, stream>>>(
      enc, w_w2frag, w_decp, w_conv, W3w, w_bias, vvec, w_scores);
  k_softmax<<<BATCH, 256, 0, stream>>>(w_scores, w_attn, o_attn);
  k_context<<<BATCH, HDIM, 0, stream>>>(w_attn, enc, w_dec, w_concat);
  k_out_gemm<<<dim3(10, 4), 256, 0, stream>>>(w_concat, w_outfrag, outb, w_logits);
  k_logsoftmax<<<BATCH, 256, 0, stream>>>(w_logits, o_logp);
}
